// leaky_integrator_current_94489280738
// MI455X (gfx1250) — matrix operand order repaired and checked by reading, not compiled, not run on hardware
//
#include <hip/hip_runtime.h>
#include <cstdint>

// ---------------- CDNA5 WMMA vector types ----------------
typedef __attribute__((ext_vector_type(16))) __bf16 v16bf;
typedef __attribute__((ext_vector_type(16))) short  v16s;
typedef __attribute__((ext_vector_type(8)))  float  v8f;

// ---------------- problem tiling constants ----------------
// n = 300 padded to M 304 (19 tiles of 16), K 320 (10 steps of 32).
#define MT   19          // M tiles of 16
#define KT   10          // K steps of 32
#define KP   320         // padded K
#define TW   64          // T-tile width (4 N-tiles of 16)
#define LDSW 336         // LDS row stride in bf16 elems (672 B = 32B multiple)
#define NWAVE 8

// f32 -> bf16 round-to-nearest-even, returned as raw u16 pattern
__device__ __host__ inline unsigned short f2bf(float f) {
    union { float f; uint32_t u; } v; v.f = f;
    uint32_t u = v.u;
    uint32_t r = u + 0x7fffu + ((u >> 16) & 1u);
    return (unsigned short)(r >> 16);
}
__device__ inline float bf2f(unsigned short h) {
    union { uint32_t u; float f; } v; v.u = ((uint32_t)h) << 16;
    return v.f;
}

// ================= prep: pack W_c / W_e into WMMA A-fragment layout =========
// A-frag (16x32 bf16): lane l holds row M = l%16; vector elem e=2v+p maps to
// K = 16*(l>=16) + e   (contiguous K order, matching B fragment loads)
__global__ void lic_pack_w(const float* __restrict__ mc,
                           const float* __restrict__ me,
                           const float* __restrict__ sc,
                           const float* __restrict__ se,
                           unsigned short* __restrict__ wcf,
                           unsigned short* __restrict__ wef,
                           int n) {
    int gid = blockIdx.x * blockDim.x + threadIdx.x;
    const int perMat = MT * KT * 32;
    if (gid >= 2 * perMat) return;
    int mat  = gid / perMat;
    int rem  = gid % perMat;
    int lane = rem & 31;
    int fi   = rem >> 5;              // mt*KT + kt
    int mt   = fi / KT;
    int kt   = fi % KT;
    int i    = mt * 16 + (lane & 15);
    unsigned short* dst = (mat ? wef : wcf) + ((size_t)fi * 32 + lane) * 16;
#pragma unroll
    for (int e = 0; e < 16; ++e) {
        int v = e >> 1, p = e & 1;
        int K = (((lane >> 4) & 1) << 4) + (v << 1) + p;
        int j = kt * 32 + K;
        float w = 0.0f;
        if (i < n && j < n) {
            if (mat == 0) w = sc[i * n + j] * mc[i * n + j] * 10.0f;
            else          w = se[i * n + j] * (me[i * n + j] + me[j * n + i]) * 0.01f;
        }
        dst[e] = f2bf(w);
    }
}

// ================= prep: rowsum(W_e) (bf16-consistent) and dt/tau ===========
__global__ void lic_prep_vec(const float* __restrict__ me,
                             const float* __restrict__ se,
                             const float* __restrict__ tau,
                             float* __restrict__ rowsum,
                             float* __restrict__ dttau,
                             int n) {
    int i = blockIdx.x * blockDim.x + threadIdx.x;
    if (i >= n) return;
    float s = 0.0f;
    for (int j = 0; j < n; ++j) {
        float w = se[i * n + j] * (me[i * n + j] + me[j * n + i]) * 0.01f;
        s += bf2f(f2bf(w));           // match bf16-rounded weights used in GEMM
    }
    rowsum[i] = s;
    dttau[i]  = 0.01f / tau[i];       // DT = 0.01
}

// packed bf16 relu: mask = per-halfword arithmetic sign-fill (v_pk_ashrrev_i16),
// result = x & ~mask (single bitop3) -> 2 VALU ops per dword, depth 2
__device__ inline v16bf relu16(v16bf x) {
    union { v16bf b; v16s s; } c; c.b = x;
    v16s m = c.s >> 15;
    c.s &= ~m;
    return c.b;
}

// ================= main GEMM + epilogue =====================================
__launch_bounds__(256)
__global__ void lic_main(const float* __restrict__ inp,
                         const float* __restrict__ hid,
                         const float* __restrict__ bias,
                         const unsigned short* __restrict__ wcf,
                         const unsigned short* __restrict__ wef,
                         const float* __restrict__ rowsum,
                         const float* __restrict__ dttau,
                         float* __restrict__ out,
                         int n, int T, int Bsz) {
    // LDS: delayed x, transposed [t_local][k] as bf16 (K contiguous -> B frags
    // load as one 32B chunk per lane; 672B row stride keeps 32B alignment)
    __shared__ unsigned short xT[TW * LDSW];

    const int b  = blockIdx.x;
    const int t0 = blockIdx.y * TW;
    const float* hb = hid + (size_t)b * n * T;
    const size_t S  = (size_t)Bsz * n * T;

    // stage + transpose + shift (x[:,t] = hidden[:,t-1], zero at t==0)
    for (int e = threadIdx.x; e < KP * TW; e += 256) {
        int j  = e / TW;
        int tl = e % TW;              // consecutive tid -> consecutive t: coalesced
        int t  = t0 + tl - 1;
        float v = 0.0f;
        if (j < n && t >= 0 && t < T) v = hb[(size_t)j * T + t];
        xT[tl * LDSW + j] = f2bf(v);
    }
    __syncthreads();

    const int wave = threadIdx.x >> 5;   // wave32 (gfx1250)
    const int lane = threadIdx.x & 31;
    const int lmod = lane & 15;
    const int lhi  = lane >> 4;

    for (int mt = wave; mt < MT; mt += NWAVE) {
        v8f accC[4] = {};                // chem accumulators, 4 N-tiles
        v8f accE[4] = {};                // elec (linear part) accumulators

        // per-(mt,lane) A-fragment base pointers; stride per k-step = 512 elems
        const unsigned short* wcBase = wcf + ((size_t)mt * KT * 32 + lane) * 16;
        const unsigned short* weBase = wef + ((size_t)mt * KT * 32 + lane) * 16;

        // software double-buffer for A fragments (hide L2 latency behind WMMAs)
        v16bf aC = *(const v16bf*)wcBase;
        v16bf aE = *(const v16bf*)weBase;

        for (int k = 0; k < KT; ++k) {
            const int kn = (k + 1 < KT) ? (k + 1) : k;
            v16bf naC = *(const v16bf*)(wcBase + (size_t)kn * 512);
            v16bf naE = *(const v16bf*)(weBase + (size_t)kn * 512);

            const int kcol = k * 32 + lhi * 16;  // lanes 0-15: K 0..15, 16-31: 16..31
            v16bf Bx[4], Br[4];
#pragma unroll
            for (int nt = 0; nt < 4; ++nt)
                Bx[nt] = *(const v16bf*)&xT[(nt * 16 + lmod) * LDSW + kcol];
#pragma unroll
            for (int nt = 0; nt < 4; ++nt)
                Br[nt] = relu16(Bx[nt]);
#pragma unroll
            for (int nt = 0; nt < 4; ++nt) {
                accC[nt] = __builtin_amdgcn_wmma_f32_16x16x32_bf16(
                    false, aC, false, Br[nt], (short)0, accC[nt], false, false);
                accE[nt] = __builtin_amdgcn_wmma_f32_16x16x32_bf16(
                    false, aE, false, Bx[nt], (short)0, accE[nt], false, false);
            }
            aC = naC;
            aE = naE;
        }

        // epilogue: C/D layout — VGPR r, lanes 0-15: M=r, N=lane; 16-31: M=8+r
        const int iBase = mt * 16 + lhi * 8;
#pragma unroll
        for (int r = 0; r < 8; ++r) {
            const int i = iBase + r;
            if (i < n) {
                const float bi  = bias[i];
                const float rs  = rowsum[i];
                const float dtt = dttau[i];
                const float* hrow = hb + (size_t)i * T;
                const size_t rowbase = ((size_t)b * n + i) * T;
#pragma unroll
                for (int nt = 0; nt < 4; ++nt) {
                    const int t = t0 + nt * 16 + lmod;
                    if (t < T) {
                        const float xv   = (t > 0) ? hrow[t - 1] : 0.0f;
                        const float chem = accC[nt][r];
                        const float elec = accE[nt][r] - rs * xv;
                        const float mu   = dtt * (chem + elec + bi + inp[rowbase + t] - xv) + xv;
                        out[rowbase + t]         = mu;    // output 0: mu
                        out[S + rowbase + t]     = chem;  // output 1: chem_in
                        out[2 * S + rowbase + t] = elec;  // output 2: elec_in
                    }
                }
            }
        }
    }
}

// ================= launcher =================================================
extern "C" void kernel_launch(void* const* d_in, const int* in_sizes, int n_in,
                              void* d_out, int out_size, void* d_ws, size_t ws_size,
                              hipStream_t stream) {
    const float* inp  = (const float*)d_in[0];
    const float* hid  = (const float*)d_in[1];
    const float* mc   = (const float*)d_in[2];
    const float* me   = (const float*)d_in[3];
    const float* sc   = (const float*)d_in[4];
    const float* se   = (const float*)d_in[5];
    const float* bias = (const float*)d_in[6];
    const float* tau  = (const float*)d_in[7];

    const int n   = in_sizes[6];            // 300
    const int Bsz = 16;                     // reference batch
    const int T   = in_sizes[0] / (Bsz * n);

    // workspace layout
    char* ws = (char*)d_ws;
    unsigned short* wcf = (unsigned short*)ws;                 // 194560 B used
    unsigned short* wef = (unsigned short*)(ws + 196608);
    float* rowsum = (float*)(ws + 2 * 196608);
    float* dttau  = (float*)(ws + 2 * 196608 + 2048);
    (void)ws_size; (void)n_in; (void)out_size;

    const int packThreads = 2 * MT * KT * 32;   // 12160
    lic_pack_w<<<(packThreads + 255) / 256, 256, 0, stream>>>(mc, me, sc, se, wcf, wef, n);
    lic_prep_vec<<<(n + 255) / 256, 256, 0, stream>>>(me, se, tau, rowsum, dttau, n);

    dim3 grid(Bsz, (T + TW - 1) / TW);          // (16, 32)
    lic_main<<<grid, 256, 0, stream>>>(inp, hid, bias, wcf, wef, rowsum, dttau,
                                       (float*)d_out, n, T, Bsz);
}
